// VisionAttention_19756849561608
// MI455X (gfx1250) — matrix operand order repaired and checked by reading, not compiled, not run on hardware
//
#include <hip/hip_runtime.h>
#include <hip/hip_bf16.h>

typedef __attribute__((ext_vector_type(16))) __bf16 v16bf;
typedef __attribute__((ext_vector_type(8)))  float  v8f;

#define SEQ 16384
#define HID 1280
#define HEADS 16
#define HEAD_DIM 80
#define NUM_WIN 16
#define WIN 1024
#define DPAD 96      // head_dim padded to 3*32 for clean WMMA K-steps
#define ANT 64       // keys per attention inner tile

static __device__ __forceinline__ unsigned short f2bf(float f) {
  unsigned int u = __float_as_uint(f);
  unsigned int r = u + 0x7FFFu + ((u >> 16) & 1u);   // round-to-nearest-even
  return (unsigned short)(r >> 16);
}
static __device__ __forceinline__ float bf2f(unsigned short s) {
  return __uint_as_float(((unsigned int)s) << 16);
}

union BF16Frag { v16bf v; unsigned short u[16]; uint4 q[2]; };

// ---------------------------------------------------------------- convert
__global__ void cvt_f32_bf16(const float* __restrict__ in,
                             unsigned short* __restrict__ out, int n) {
  int i = blockIdx.x * blockDim.x + threadIdx.x;
  if (i < n) out[i] = f2bf(in[i]);
}

// ---------------------------------------------------------------- GEMM
// C[M,N] = A[M,K](bf16) @ B[K,N](bf16) + bias.
// Block 128x128, 8 waves (2x4), wave tile 64x32 -> 4x2 WMMA 16x16x32 tiles.
// LDS double-buffered; B tile stored transposed [n][k] so every fragment is
// two contiguous ds_load_b128s.
template <int OUT_BF16>
__global__ __launch_bounds__(256) void gemm_bf16_wmma(
    const unsigned short* __restrict__ A, const unsigned short* __restrict__ B,
    const float* __restrict__ bias, void* __restrict__ C,
    int M, int N, int K)
{
  __shared__ __align__(16) unsigned short As[2][128][40];  // [m][k], padded
  __shared__ __align__(16) unsigned short Bt[2][128][40];  // [n][k], padded

  const int tid  = threadIdx.x;
  const int lane = tid & 31;
  const int wave = tid >> 5;
  const int half = lane >> 4, ln = lane & 15;
  const int wm0 = (wave >> 2) * 64;     // 0 or 64
  const int wn0 = (wave & 3) * 32;      // 0,32,64,96
  const int bm = blockIdx.y * 128, bn = blockIdx.x * 128;

  const int arow = tid >> 1, acol = (tid & 1) * 16;
  const int brow = tid >> 3, bcol = (tid & 7) * 16;

  v8f zero = {};
  v8f acc[4][2];
#pragma unroll
  for (int i = 0; i < 4; ++i)
#pragma unroll
    for (int j = 0; j < 2; ++j) acc[i][j] = zero;

  const unsigned short* Ab = A + (size_t)(bm + arow) * K + acol;
  const unsigned short* Bb = B + (size_t)brow * N + bn + bcol;

  // ---- prologue: stage tile 0 into buffer 0
  {
    uint4 a0 = *(const uint4*)(Ab);
    uint4 a1 = *(const uint4*)(Ab + 8);
    uint4 b0 = *(const uint4*)(Bb);
    uint4 b1 = *(const uint4*)(Bb + 8);
    *(uint4*)&As[0][arow][acol]     = a0;
    *(uint4*)&As[0][arow][acol + 8] = a1;
    unsigned short btmp[16];
    *(uint4*)&btmp[0] = b0;
    *(uint4*)&btmp[8] = b1;
#pragma unroll
    for (int j = 0; j < 16; ++j) Bt[0][bcol + j][brow] = btmp[j];
  }
  __syncthreads();

  const int nT = K >> 5;
  for (int t = 0; t < nT; ++t) {
    const int cur = t & 1;
    uint4 a0, a1, b0, b1;
    const bool more = (t + 1) < nT;
    if (more) {
      const unsigned short* Ap = Ab + (t + 1) * 32;
      a0 = *(const uint4*)(Ap);
      a1 = *(const uint4*)(Ap + 8);
      const unsigned short* Bp = Bb + (size_t)(t + 1) * 32 * N;
      b0 = *(const uint4*)(Bp);
      b1 = *(const uint4*)(Bp + 8);
    }

    BF16Frag af[4], bfr[2];
#pragma unroll
    for (int mi = 0; mi < 4; ++mi) {
      const unsigned short* p = &As[cur][wm0 + mi * 16 + ln][0];
      af[mi].q[0] = *(const uint4*)(p + 8 * half);
      af[mi].q[1] = *(const uint4*)(p + 16 + 8 * half);
    }
#pragma unroll
    for (int ni = 0; ni < 2; ++ni) {
      const unsigned short* p = &Bt[cur][wn0 + ni * 16 + ln][0];
      bfr[ni].q[0] = *(const uint4*)(p + 8 * half);
      bfr[ni].q[1] = *(const uint4*)(p + 16 + 8 * half);
    }
#pragma unroll
    for (int mi = 0; mi < 4; ++mi)
#pragma unroll
      for (int ni = 0; ni < 2; ++ni)
        acc[mi][ni] = __builtin_amdgcn_wmma_f32_16x16x32_bf16(
            false, af[mi].v, false, bfr[ni].v, (short)0, acc[mi][ni], false, false);

    if (more) {
      const int nxt = cur ^ 1;
      *(uint4*)&As[nxt][arow][acol]     = a0;
      *(uint4*)&As[nxt][arow][acol + 8] = a1;
      unsigned short btmp[16];
      *(uint4*)&btmp[0] = b0;
      *(uint4*)&btmp[8] = b1;
#pragma unroll
      for (int j = 0; j < 16; ++j) Bt[nxt][bcol + j][brow] = btmp[j];
      __syncthreads();
    }
  }

#pragma unroll
  for (int mi = 0; mi < 4; ++mi)
#pragma unroll
    for (int ni = 0; ni < 2; ++ni)
#pragma unroll
      for (int r = 0; r < 8; ++r) {
        int row = bm + wm0 + mi * 16 + r + 8 * half;
        int col = bn + wn0 + ni * 16 + ln;
        float val = acc[mi][ni][r] + bias[col];
        if (OUT_BF16) ((unsigned short*)C)[(size_t)row * N + col] = f2bf(val);
        else          ((float*)C)[(size_t)row * N + col] = val;
      }
}

// ---------------------------------------------------------------- RoPE + pack
// qkv bf16 [S][3*HID] -> Q/K/V bf16 [H][S][96] (pad 80..95 = 0); Q pre-scaled.
__global__ void rope_pack(const unsigned short* __restrict__ qkv,
                          const float* __restrict__ cosb, const float* __restrict__ sinb,
                          unsigned short* __restrict__ Qp, unsigned short* __restrict__ Kp,
                          unsigned short* __restrict__ Vp)
{
  int i = blockIdx.x * blockDim.x + threadIdx.x;
  int d = i % DPAD;
  int h = (i / DPAD) % HEADS;
  int s = i / (DPAD * HEADS);
  if (s >= SEQ) return;
  size_t o = ((size_t)h * SEQ + s) * DPAD + d;
  if (d >= HEAD_DIM) { Qp[o] = 0; Kp[o] = 0; Vp[o] = 0; return; }
  size_t qb = (size_t)s * (3 * HID) + h * HEAD_DIM + d;
  float c  = cosb[s * HEAD_DIM + d];
  float sn = sinb[s * HEAD_DIM + d];
  int dp = (d < HEAD_DIM / 2) ? d + HEAD_DIM / 2 : d - HEAD_DIM / 2;
  float sgn = (d < HEAD_DIM / 2) ? -1.f : 1.f;
  size_t pb = (size_t)s * (3 * HID) + h * HEAD_DIM + dp;
  float q  = bf2f(qkv[qb]),       qp = bf2f(qkv[pb]);
  float k  = bf2f(qkv[qb + HID]), kp = bf2f(qkv[pb + HID]);
  const float scale = 0.11180339887498949f;    // 80^-0.5 folded into Q
  Qp[o] = f2bf((q * c + sgn * qp * sn) * scale);
  Kp[o] = f2bf(k * c + sgn * kp * sn);
  Vp[o] = qkv[qb + 2 * HID];
}

// ---------------------------------------------------------------- attention
// grid (WIN/128, HEADS, NUM_WIN); 8 waves, each owns 16 query rows.
// K tile [key][dim] (dim contiguous -> b128 frags for Q.K^T);
// V tile stored transposed [dim][key] -> b128 frags for P.V.
__global__ __launch_bounds__(256) void attn_wmma(
    const unsigned short* __restrict__ Qp, const unsigned short* __restrict__ Kp,
    const unsigned short* __restrict__ Vp, unsigned short* __restrict__ Out)
{
  __shared__ __align__(16) unsigned short Ks[ANT][104];    // [key][dim], padded
  __shared__ __align__(16) unsigned short Vt[DPAD][72];    // [dim][key], padded
  __shared__ __align__(16) unsigned short Ps[8][16][ANT];  // per-wave P

  const int tid = threadIdx.x, wave = tid >> 5, lane = tid & 31;
  const int half = lane >> 4, ln = lane & 15;
  const int head = blockIdx.y, win = blockIdx.z;
  const int srow0 = win * WIN + blockIdx.x * 128 + wave * 16;
  const size_t hbase = (size_t)head * SEQ;

  BF16Frag qf[3];                       // Q rows (16x96) as 3 A-fragments
  {
    const unsigned short* qptr = Qp + (hbase + srow0 + ln) * DPAD;
#pragma unroll
    for (int f = 0; f < 3; ++f) {
      qf[f].q[0] = *(const uint4*)(qptr + f * 32 + 8 * half);
      qf[f].q[1] = *(const uint4*)(qptr + f * 32 + 16 + 8 * half);
    }
  }

  float mrow[8], lrow[8];
  v8f zero = {};
  v8f oacc[6];
#pragma unroll
  for (int r = 0; r < 8; ++r) { mrow[r] = -3.0e38f; lrow[r] = 0.f; }
#pragma unroll
  for (int t = 0; t < 6; ++t) oacc[t] = zero;

  for (int kb0 = 0; kb0 < WIN; kb0 += ANT) {
    __syncthreads();
#pragma unroll
    for (int c = 0; c < 3; ++c) {                 // 64x96 K and V tiles
      int e = (tid * 3 + c) * 8;
      int row = e / DPAD, col = e % DPAD;
      size_t g = (hbase + win * WIN + kb0 + row) * DPAD + col;
      *(uint4*)&Ks[row][col] = *(const uint4*)(Kp + g);
      unsigned short vtmp[8];
      *(uint4*)&vtmp[0] = *(const uint4*)(Vp + g);
#pragma unroll
      for (int j = 0; j < 8; ++j) Vt[col + j][row] = vtmp[j];
    }
    __syncthreads();

    v8f sacc[4];                                  // 16x64 scores
#pragma unroll
    for (int j = 0; j < 4; ++j) {
      sacc[j] = zero;
#pragma unroll
      for (int f = 0; f < 3; ++f) {
        BF16Frag bk;                              // B = K^T (dim x key)
        const unsigned short* p = &Ks[j * 16 + ln][f * 32];
        bk.q[0] = *(const uint4*)(p + 8 * half);
        bk.q[1] = *(const uint4*)(p + 16 + 8 * half);
        sacc[j] = __builtin_amdgcn_wmma_f32_16x16x32_bf16(
            false, qf[f].v, false, bk.v, (short)0, sacc[j], false, false);
      }
    }
    // online softmax (row r on this lane == query row r + 8*half)
#pragma unroll
    for (int r = 0; r < 8; ++r) {
      float mx = sacc[0][r];
#pragma unroll
      for (int j = 1; j < 4; ++j) mx = fmaxf(mx, sacc[j][r]);
#pragma unroll
      for (int m = 1; m < 16; m <<= 1) mx = fmaxf(mx, __shfl_xor(mx, m, 32));
      float nm = fmaxf(mrow[r], mx);
      float alpha = __expf(mrow[r] - nm);
      mrow[r] = nm;
      float ps = 0.f;
#pragma unroll
      for (int j = 0; j < 4; ++j) {
        float p = __expf(sacc[j][r] - nm);
        sacc[j][r] = p;
        ps += p;
      }
#pragma unroll
      for (int m = 1; m < 16; m <<= 1) ps += __shfl_xor(ps, m, 32);
      lrow[r] = lrow[r] * alpha + ps;
#pragma unroll
      for (int t = 0; t < 6; ++t) oacc[t][r] *= alpha;
      int prow = r + 8 * half;
#pragma unroll
      for (int j = 0; j < 4; ++j)
        Ps[wave][prow][j * 16 + ln] = f2bf(sacc[j][r]);
    }
    // P(16xANT) @ V(ANTx96): 2 K-steps x 6 N-tiles
#pragma unroll
    for (int f = 0; f < 2; ++f) {
      BF16Frag pa;
      const unsigned short* pp = &Ps[wave][ln][f * 32];
      pa.q[0] = *(const uint4*)(pp + 8 * half);
      pa.q[1] = *(const uint4*)(pp + 16 + 8 * half);
#pragma unroll
      for (int t = 0; t < 6; ++t) {
        BF16Frag vb;
        const unsigned short* vp = &Vt[t * 16 + ln][f * 32];
        vb.q[0] = *(const uint4*)(vp + 8 * half);
        vb.q[1] = *(const uint4*)(vp + 16 + 8 * half);
        oacc[t] = __builtin_amdgcn_wmma_f32_16x16x32_bf16(
            false, pa.v, false, vb.v, (short)0, oacc[t], false, false);
      }
    }
  }

#pragma unroll
  for (int t = 0; t < 5; ++t)                     // cols 80..95 are padding
#pragma unroll
    for (int r = 0; r < 8; ++r) {
      int row = r + 8 * half;
      int s = srow0 + row;
      int dcol = t * 16 + ln;
      float val = oacc[t][r] / lrow[r];
      Out[(size_t)s * HID + head * HEAD_DIM + dcol] = f2bf(val);
    }
}

// ---------------------------------------------------------------- launch
extern "C" void kernel_launch(void* const* d_in, const int* in_sizes, int n_in,
                              void* d_out, int out_size, void* d_ws, size_t ws_size,
                              hipStream_t stream)
{
  (void)in_sizes; (void)n_in; (void)out_size; (void)ws_size;
  const float* hs    = (const float*)d_in[0];
  const float* cosb  = (const float*)d_in[1];
  const float* sinb  = (const float*)d_in[2];
  const float* wqkv  = (const float*)d_in[3];
  const float* bqkv  = (const float*)d_in[4];
  const float* wproj = (const float*)d_in[5];
  const float* bproj = (const float*)d_in[6];
  // d_in[7] = cu_seqlens: windows are uniform (SEQ/NUM_WIN), layout baked in.

  char* ws = (char*)d_ws;
  size_t off = 0;
  auto alloc = [&](size_t bytes) -> char* {
    char* p = ws + off;
    off += (bytes + 255) & ~(size_t)255;
    return p;
  };
  unsigned short* hs_bf    = (unsigned short*)alloc((size_t)SEQ * HID * 2);
  unsigned short* wqkv_bf  = (unsigned short*)alloc((size_t)HID * 3 * HID * 2);
  unsigned short* wproj_bf = (unsigned short*)alloc((size_t)HID * HID * 2);
  unsigned short* qkv_bf   = (unsigned short*)alloc((size_t)SEQ * 3 * HID * 2);
  unsigned short* Qp       = (unsigned short*)alloc((size_t)HEADS * SEQ * DPAD * 2);
  unsigned short* Kp       = (unsigned short*)alloc((size_t)HEADS * SEQ * DPAD * 2);
  unsigned short* Vp       = (unsigned short*)alloc((size_t)HEADS * SEQ * DPAD * 2);
  unsigned short* attn_bf  = (unsigned short*)alloc((size_t)SEQ * HID * 2);

  int n1 = SEQ * HID;
  cvt_f32_bf16<<<(n1 + 255) / 256, 256, 0, stream>>>(hs, hs_bf, n1);
  int n2 = HID * 3 * HID;
  cvt_f32_bf16<<<(n2 + 255) / 256, 256, 0, stream>>>(wqkv, wqkv_bf, n2);
  int n3 = HID * HID;
  cvt_f32_bf16<<<(n3 + 255) / 256, 256, 0, stream>>>(wproj, wproj_bf, n3);

  gemm_bf16_wmma<1><<<dim3((3 * HID) / 128, SEQ / 128), 256, 0, stream>>>(
      hs_bf, wqkv_bf, bqkv, qkv_bf, SEQ, 3 * HID, HID);

  int n4 = SEQ * HEADS * DPAD;
  rope_pack<<<(n4 + 255) / 256, 256, 0, stream>>>(qkv_bf, cosb, sinb, Qp, Kp, Vp);

  attn_wmma<<<dim3(WIN / 128, HEADS, NUM_WIN), 256, 0, stream>>>(Qp, Kp, Vp, attn_bf);

  gemm_bf16_wmma<0><<<dim3(HID / 128, SEQ / 128), 256, 0, stream>>>(
      attn_bf, wproj_bf, bproj, d_out, SEQ, HID, HID);
}
